// GSchNetContinuousFilterConvolution_45767171506190
// MI455X (gfx1250) — matrix operand order repaired and checked by reading, not compiled, not run on hardware
//
#include <hip/hip_runtime.h>
#include <math.h>

typedef __attribute__((ext_vector_type(16))) _Float16 v16h;
typedef __attribute__((ext_vector_type(8)))  _Float16 v8h;
typedef __attribute__((ext_vector_type(8)))  float    v8f;

#define LATENT 128
#define NCENT  100

__global__ __launch_bounds__(256)
void gschnet_cfconv_wmma(const float* __restrict__ nodes,
                         const int*   __restrict__ senders,
                         const int*   __restrict__ receivers,
                         const float* __restrict__ W1,
                         const float* __restrict__ b1,
                         const float* __restrict__ W2,
                         const float* __restrict__ b2,
                         float*       __restrict__ out,
                         int nEdges)
{
    // A-fragment staging (RBF tile, 16 edges x 128 padded centers) in A-layout:
    // [kstep][lane][half] -> one v16h per lane per kstep.
    __shared__ __align__(32) _Float16 aFrag[4][32][16];       // 4 KB
    // h tile (16 edges x 128 latent), row-major f16 for A-layout reloads.
    __shared__ __align__(32) _Float16 hBuf[16][LATENT];       // 4 KB

    const int tid    = threadIdx.x;
    const int lane   = tid & 31;
    const int wave   = tid >> 5;              // 0..7: owns output n-tile `wave`
    const int col    = lane & 15;             // N within tile / edge-row for A
    const int hiHalf = (lane & 16) ? 1 : 0;

    // ---- persistent B-fragments of W1 (K padded 100->128) and W2 ----
    // B layout (16-bit 32x16), same K order as A:
    // half h -> K = kk*32 + (h<8 ? h : h+8) + (hiHalf ? 8 : 0), N = col.
    const int n = wave * 16 + col;
    v16h w1f[4], w2f[4];
    #pragma unroll
    for (int kk = 0; kk < 4; ++kk) {
        #pragma unroll
        for (int h = 0; h < 16; ++h) {
            int k = kk * 32 + ((h < 8) ? h : h + 8) + (hiHalf ? 8 : 0);
            float v1 = (k < NCENT) ? W1[k * LATENT + n] : 0.0f;
            float v2 = W2[k * LATENT + n];
            w1f[kk][h] = (_Float16)v1;
            w2f[kk][h] = (_Float16)v2;
        }
    }
    const float bb1 = b1[n];
    const float bb2 = b2[n];

    const int   numTiles = nEdges >> 4;       // E divisible by 16
    const float cstep    = 10.0f / 99.0f;     // linspace(0,10,100) spacing

    for (int tile = blockIdx.x; tile < numTiles; tile += gridDim.x) {
        // ---- phase 0: waves 0..3 build RBF A-fragment for kstep = wave ----
        if (wave < 4) {
            int   e  = tile * 16 + col;       // edge row handled by this lane
            int   s  = senders[e];
            int   r  = receivers[e];
            float dx = nodes[3 * s + 0] - nodes[3 * r + 0];
            float dy = nodes[3 * s + 1] - nodes[3 * r + 1];
            float dz = nodes[3 * s + 2] - nodes[3 * r + 2];
            float d  = sqrtf(dx * dx + dy * dy + dz * dz);
            // A layout (16-bit 16x32): halves 0..7 -> K base+0..7 (lo) / +8..15 (hi),
            // halves 8..15 -> K base+16..23 (lo) / +24..31 (hi).
            v16h a;
            #pragma unroll
            for (int h = 0; h < 16; ++h) {
                int   k = wave * 32 + ((h < 8) ? h : h + 8) + (hiHalf ? 8 : 0);
                float c = (float)k * cstep;
                float t = d - c;
                float v = (k < NCENT) ? __expf(-t * t) : 0.0f;
                a[h] = (_Float16)v;
            }
            *(v16h*)&aFrag[wave][lane][0] = a;
        }
        __syncthreads();

        // ---- layer 1: this wave's 16 columns of h = ssp(rbf @ W1 + b1) ----
        v8f acc = {};
        #pragma unroll
        for (int kk = 0; kk < 4; ++kk) {
            v16h a = *(const v16h*)&aFrag[kk][lane][0];
            acc = __builtin_amdgcn_wmma_f32_16x16x32_f16(
                      false, a, false, w1f[kk], (short)0, acc, false, false);
        }
        // C layout: element r of lane -> row M = r + 8*hiHalf, col N = n.
        #pragma unroll
        for (int r = 0; r < 8; ++r) {
            float x  = acc[r] + bb1;
            // shifted softplus, numerically stable: max(x,0)+log1p(exp(-|x|))-ln2
            float sp = fmaxf(x, 0.0f)
                     + __logf(1.0f + __expf(-fabsf(x)))
                     - 0.69314718056f;
            int row = r + (hiHalf ? 8 : 0);
            hBuf[row][n] = (_Float16)sp;
        }
        __syncthreads();

        // ---- layer 2: out n-tile = h @ W2 + b2 ----
        v8f acc2 = {};
        {
            const int rowA = col;             // A row = edge within tile
            #pragma unroll
            for (int kk = 0; kk < 4; ++kk) {
                int c0 = kk * 32 + (hiHalf ? 8 : 0);
                v8h lo = *(const v8h*)&hBuf[rowA][c0];        // K base+0..7
                v8h hi = *(const v8h*)&hBuf[rowA][c0 + 16];   // K base+16..23
                v16h a = __builtin_shufflevector(lo, hi,
                          0, 1, 2, 3, 4, 5, 6, 7, 8, 9, 10, 11, 12, 13, 14, 15);
                acc2 = __builtin_amdgcn_wmma_f32_16x16x32_f16(
                           false, a, false, w2f[kk], (short)0, acc2, false, false);
            }
        }
        #pragma unroll
        for (int r = 0; r < 8; ++r) {
            int row = r + (hiHalf ? 8 : 0);
            int e   = tile * 16 + row;
            out[(size_t)e * LATENT + n] = acc2[r] + bb2;
        }
        __syncthreads();   // protect aFrag/hBuf before next tile overwrites
    }
}

extern "C" void kernel_launch(void* const* d_in, const int* in_sizes, int n_in,
                              void* d_out, int out_size, void* d_ws, size_t ws_size,
                              hipStream_t stream) {
    (void)n_in; (void)out_size; (void)d_ws; (void)ws_size;
    const float* nodes     = (const float*)d_in[0];
    const int*   senders   = (const int*)  d_in[1];
    const int*   receivers = (const int*)  d_in[2];
    const float* W1        = (const float*)d_in[3];
    const float* b1        = (const float*)d_in[4];
    const float* W2        = (const float*)d_in[5];
    const float* b2        = (const float*)d_in[6];
    float*       out       = (float*)d_out;

    int nEdges   = in_sizes[1];
    int numTiles = nEdges >> 4;
    int grid     = numTiles < 2500 ? (numTiles > 0 ? numTiles : 1) : 2500;

    gschnet_cfconv_wmma<<<grid, 256, 0, stream>>>(
        nodes, senders, receivers, W1, b1, W2, b2, out, nEdges);
}
